// InferencePipeline_6107443495378
// MI455X (gfx1250) — matrix operand order repaired and checked by reading, not compiled, not run on hardware
//
#include <hip/hip_runtime.h>
#include <hip/hip_bf16.h>

typedef unsigned int u32;
typedef __attribute__((ext_vector_type(16))) _Float16 v16h;
typedef __attribute__((ext_vector_type(8)))  float    v8f;
typedef __attribute__((ext_vector_type(4)))  unsigned int v4u;
typedef __attribute__((ext_vector_type(8)))  int v8i;
typedef __attribute__((ext_vector_type(4)))  int v4i;

#define CT 80
#define CS 54
#define DD 256
#define HH 256
#define WW 512
#define HW (HH*WW)            /* 131072 */
#define KTOP 100
#define CAP 4096
#define CENTER_THRES 0.1f
#define SEM_THRES 0.1f

/* ---- workspace byte offsets ---- */
#define WS_HIST   0            /* 65536 u32 = 262144 B */
#define WS_CNT    262144       /* 1 u32 candidate counter */
#define WS_CUT    262148       /* 1 u32 cutoff bin */
#define WS_SCNT   262400       /* 64 u32 stuff counts */
#define WS_SSUM   262656       /* 64 f32 stuff score sums */
#define WS_SPAT   262912       /* 128 i32 spatial idx of top-K */
#define WS_CANDV  263424       /* 4096 f32 */
#define WS_CANDI  279808       /* 4096 i32 */
#define WS_ACC    296192       /* 64*256 f32 GEMM accum */
#define WS_CATS   361728       /* HW i32 argmax classes */
#define WS_ZERO_WORDS (361728/4)

/* ---- output element offsets (floats) ---- */
#define OUT_TK    0            /* [100,256] thing kernels  */
#define OUT_TS    25600        /* [100] thing scores       */
#define OUT_TC    25700        /* [100] thing cats         */
#define OUT_TKEEP 25800        /* [100] thing keep         */
#define OUT_SK    25900        /* [54,256] stuff kernels   */
#define OUT_SS    39724        /* [54] stuff scores        */
#define OUT_SKEEP 39778        /* [54] stuff keep          */

/* ---- stuff GEMM config ---- */
#define PCHUNK    64           /* pixels per TDM tile */
#define TSTRIDE   65           /* LDS row stride (TDM pad: 64DW data + 1DW pad) */
#define S2_GRID   256
#define S2_CPB    (HW / PCHUNK / S2_GRID)   /* 8 chunks per block */

__device__ __forceinline__ float sigmoidf_(float x) {
    return 1.0f / (1.0f + __expf(-x));
}

/* =================== K0: zero workspace =================== */
__global__ void zero_ws_kernel(u32* w, int n) {
    int i = blockIdx.x * 256 + threadIdx.x;
    if (i < n) w[i] = 0u;
}

/* =================== T1/K3: peak detection ===================
 * phase 0: histogram of top-16 float bits of masked peak values > thres
 * phase 1: compact candidates with bin >= cutoff into small list     */
__global__ void __launch_bounds__(256)
peaks_kernel(const float* __restrict__ thing, u32* hist, const u32* cut,
             u32* cand_cnt, float* cand_v, int* cand_i, int phase) {
    __shared__ float sg[36 * 36];
    __shared__ float sm[34 * 34];
    int bid = blockIdx.x;
    int c   = bid >> 7;          /* 128 tiles per channel */
    int tb  = bid & 127;
    int ty  = tb >> 4, tx = tb & 15;
    int y0  = ty * 32, x0 = tx * 32;
    const float* base = thing + (size_t)c * HW;

    for (int i = threadIdx.x; i < 36 * 36; i += 256) {
        int ly = i / 36, lx = i - ly * 36;
        int gy = y0 + ly - 2, gx = x0 + lx - 2;
        float v = 0.0f;  /* zero-pad => count_include_pad avg semantics */
        if ((unsigned)gy < (unsigned)HH && (unsigned)gx < (unsigned)WW)
            v = sigmoidf_(base[gy * WW + gx]);
        sg[i] = v;
    }
    __syncthreads();

    for (int i = threadIdx.x; i < 34 * 34; i += 256) {
        int ry = i / 34, rx = i - ry * 34;
        int ly = ry + 1, lx = rx + 1;           /* sg coords */
        int gy = y0 + ly - 2, gx = x0 + lx - 2;
        float r = -1e30f;                       /* -inf pad for maxpool */
        if ((unsigned)gy < (unsigned)HH && (unsigned)gx < (unsigned)WW) {
            float s = 0.0f;
            #pragma unroll
            for (int dy = -1; dy <= 1; ++dy)
                #pragma unroll
                for (int dx = -1; dx <= 1; ++dx)
                    s += sg[(ly + dy) * 36 + lx + dx];
            r = (sg[ly * 36 + lx] + s * (1.0f / 9.0f)) * 0.5f;
        }
        sm[i] = r;
    }
    __syncthreads();

    u32 cutv = (phase == 1) ? *cut : 0u;
    for (int i = threadIdx.x; i < 32 * 32; i += 256) {
        int iy = i >> 5, ix = i & 31;
        int my = iy + 1, mx = ix + 1;           /* sm coords */
        float v = sm[my * 34 + mx];
        float m = v;
        #pragma unroll
        for (int dy = -1; dy <= 1; ++dy)
            #pragma unroll
            for (int dx = -1; dx <= 1; ++dx)
                m = fmaxf(m, sm[(my + dy) * 34 + mx + dx]);
        if (v >= m && v > CENTER_THRES) {       /* is local max of smoothed */
            u32 b = __float_as_uint(v) >> 16;   /* monotone for v in (0,1) */
            if (phase == 0) {
                atomicAdd(&hist[b], 1u);
            } else if (b >= cutv) {
                u32 pos = atomicAdd(cand_cnt, 1u);
                if (pos < CAP) {
                    cand_v[pos] = v;
                    cand_i[pos] = c * HW + (y0 + iy) * WW + (x0 + ix);
                }
            }
        }
    }
}

/* =================== K2: find cutoff bin =================== */
__global__ void __launch_bounds__(256)
cutoff_kernel(const u32* __restrict__ hist, u32* cut) {
    __shared__ u32 part[256];
    int t = threadIdx.x;
    u32 s = 0;
    for (int i = 0; i < 256; ++i) s += hist[t * 256 + i];
    part[t] = s;
    __syncthreads();
    if (t == 0) {
        u32 acc = 0; int blk = -1;
        for (int b = 255; b >= 0; --b) {
            if (acc + part[b] >= (u32)KTOP) { blk = b; break; }
            acc += part[b];
        }
        u32 c = 0;
        if (blk >= 0) {
            u32 a2 = acc;
            for (int i = blk * 256 + 255; i >= blk * 256; --i) {
                a2 += hist[i];
                if (a2 >= (u32)KTOP) { c = (u32)i; break; }
            }
        }
        *cut = c;
    }
}

/* =================== K4: exact sorted top-100 =================== */
__global__ void __launch_bounds__(256)
select_kernel(const u32* __restrict__ cand_cnt,
              const float* __restrict__ cand_v, const int* __restrict__ cand_i,
              float* __restrict__ out, int* __restrict__ spat) {
    __shared__ float lv[CAP];
    __shared__ int   li[CAP];
    __shared__ float bv[256];
    __shared__ int   bi[256];
    __shared__ int   bp[256];
    int tid = threadIdx.x;
    int n = (int)*cand_cnt; if (n > CAP) n = CAP;
    for (int i = tid; i < CAP; i += 256) {
        lv[i] = (i < n) ? cand_v[i] : -1.0f;
        li[i] = (i < n) ? cand_i[i] : 0x7FFFFFFF;
    }
    __syncthreads();
    for (int r = 0; r < KTOP; ++r) {
        float mv = -1.0f; int mi = 0x7FFFFFFF; int mp = -1;
        for (int i = tid; i < CAP; i += 256) {
            float v = lv[i]; int id = li[i];
            if (v > mv || (v == mv && id < mi)) { mv = v; mi = id; mp = i; }
        }
        bv[tid] = mv; bi[tid] = mi; bp[tid] = mp;
        __syncthreads();
        if (tid == 0) {
            float Mv = -1.0f; int Mi = 0x7FFFFFFF, Mp = -1;
            for (int t2 = 0; t2 < 256; ++t2) {
                if (bv[t2] > Mv || (bv[t2] == Mv && bi[t2] < Mi)) {
                    Mv = bv[t2]; Mi = bi[t2]; Mp = bp[t2];
                }
            }
            if (Mp >= 0 && Mv > 0.0f) {
                lv[Mp] = -1.0f;
                out[OUT_TS + r]    = Mv;              /* all collected > thres */
                out[OUT_TC + r]    = (float)(Mi / HW);
                out[OUT_TKEEP + r] = 1.0f;
                spat[r] = Mi % HW;
            } else {
                out[OUT_TS + r] = 0.0f; out[OUT_TC + r] = 0.0f;
                out[OUT_TKEEP + r] = 0.0f; spat[r] = -1;
            }
        }
        __syncthreads();
    }
}

/* =================== K5: gather thing kernels =================== */
__global__ void __launch_bounds__(256)
gather_kernel(const float* __restrict__ ksp, const int* __restrict__ spat,
              float* __restrict__ out) {
    int r = blockIdx.x;
    int d = threadIdx.x;
    int sp = spat[r];
    out[OUT_TK + r * DD + d] = (sp >= 0) ? ksp[(size_t)d * HW + sp] : 0.0f;
}

/* =================== S1: stuff argmax + per-class stats =================== */
__global__ void __launch_bounds__(256)
stuff_argmax_kernel(const float* __restrict__ smap, int* __restrict__ cats,
                    u32* __restrict__ scnt, float* __restrict__ ssum) {
    __shared__ u32   lc[64];
    __shared__ float ls[64];
    int t = threadIdx.x;
    if (t < 64) { lc[t] = 0u; ls[t] = 0.0f; }
    __syncthreads();
    int p = blockIdx.x * 256 + t;
    float best = -1.0f; int bc = 0;
    for (int c = 0; c < CS; ++c) {
        float s = sigmoidf_(smap[(size_t)c * HW + p]);
        if (s > best) { best = s; bc = c; }     /* first max wins (argmax) */
    }
    cats[p] = bc;
    atomicAdd(&lc[bc], 1u);
    atomicAdd(&ls[bc], best);
    __syncthreads();
    if (t < CS) {
        if (lc[t]) atomicAdd(&scnt[t], lc[t]);
        atomicAdd(&ssum[t], ls[t]);
    }
}

/* =================== S2: WMMA GEMM onehot^T x kernel_space^T ===================
 * C[c,d] = sum_p onehot[c,p]*ks[d,p].  M=64 (CS pad), N=256, K=HW.
 * kernel_space tiles streamed by TDM into LDS (padded stride 65 -> no bank
 * conflicts on transposed B reads), double-buffered on TENSORcnt.           */
__device__ __forceinline__ void tdm_issue(const float* ksp, int p0, float* ldsdst) {
    unsigned long long ga = (unsigned long long)(const void*)(ksp + p0);
    unsigned int lb = (unsigned int)(unsigned long long)(void*)ldsdst;
    v4u g0;
    g0[0] = 1u;                                       /* count=1 */
    g0[1] = lb;                                       /* lds_addr */
    g0[2] = (unsigned int)ga;                         /* global_addr[31:0]  */
    g0[3] = (unsigned int)((ga >> 32) & 0x01FFFFFFull) | 0x80000000u; /* +type=2 */
    v8i g1;
    g1[0] = (int)((2u << 16) | (1u << 20) | (5u << 22)); /* 4B elems, pad 1DW/64DW */
    g1[1] = (int)(((u32)HW & 0xFFFFu) << 16);            /* tensor_dim0 lo16 */
    g1[2] = (int)((((u32)HW >> 16) & 0xFFFFu) | (((u32)DD & 0xFFFFu) << 16));
    g1[3] = (int)((((u32)DD >> 16) & 0xFFFFu) | ((u32)PCHUNK << 16)); /* tile_dim0 */
    g1[4] = (int)((u32)DD & 0xFFFFu);                    /* tile_dim1=256 */
    g1[5] = (int)HW;                                     /* dim0_stride lo32 */
    g1[6] = 0; g1[7] = 0;
    v4i z4 = (v4i)0;
    v8i z8 = (v8i)0;
    /* amdgpu-toolchain clang-23: 6-arg form (g0, g1, g2, g3, g4, cpol) */
    __builtin_amdgcn_tensor_load_to_lds(g0, g1, z4, z4, z8, 0);
}

__global__ void __launch_bounds__(256)
stuff_gemm_kernel(const float* __restrict__ ksp, const int* __restrict__ cats,
                  float* __restrict__ accw) {
    __shared__ float tile[2][DD * TSTRIDE];   /* 2 x 66560 B */
    __shared__ int   sc[2][PCHUNK];
    const int tid  = threadIdx.x;
    const int wave = tid >> 5;
    const int lane = tid & 31;
    const int hi   = lane >> 4;
    const int ln   = lane & 15;

    int c0 = blockIdx.x * S2_CPB;

    /* prologue: chunk c0 -> buffer 0 */
    if (tid < 32) tdm_issue(ksp, c0 * PCHUNK, &tile[0][0]);
    if (tid < PCHUNK) sc[0][tid] = cats[c0 * PCHUNK + tid];

    v8f acc[4][2] = {};

    for (int i = 0; i < S2_CPB; ++i) {
        int buf = i & 1;
        if (i + 1 < S2_CPB) {
            if (tid < 32) tdm_issue(ksp, (c0 + i + 1) * PCHUNK, &tile[buf ^ 1][0]);
            if (tid < PCHUNK) sc[buf ^ 1][tid] = cats[(c0 + i + 1) * PCHUNK + tid];
            if (tid < 32) __builtin_amdgcn_s_wait_tensorcnt(1);
        } else {
            if (tid < 32) __builtin_amdgcn_s_wait_tensorcnt(0);
        }
        __syncthreads();

        const float* tl = &tile[buf][0];
        const int*   cc = &sc[buf][0];
        #pragma unroll
        for (int kstep = 0; kstep < PCHUNK / 32; ++kstep) {
            int pb = kstep * 32;
            /* A fragments: onehot, 16-bit A 16x32 layout */
            v16h A[4];
            #pragma unroll
            for (int m = 0; m < 4; ++m) {
                int cls = m * 16 + ln;
                #pragma unroll
                for (int e = 0; e < 16; ++e) {
                    int k = ((e < 8) ? e : e + 8) + (hi ? 8 : 0);
                    A[m][e] = (cc[pb + k] == cls) ? (_Float16)1.0f : (_Float16)0.0f;
                }
            }
            #pragma unroll
            for (int j = 0; j < 2; ++j) {
                int nt   = wave + 8 * j;          /* N-tile (d block) */
                int drow = nt * 16 + ln;
                v16h B;
                #pragma unroll
                for (int e = 0; e < 16; ++e) {
                    int k = ((e < 8) ? e : e + 8) + (hi ? 8 : 0);
                    B[e] = (_Float16)tl[drow * TSTRIDE + pb + k];
                }
                #pragma unroll
                for (int m = 0; m < 4; ++m)
                    acc[m][j] = __builtin_amdgcn_wmma_f32_16x16x32_f16(
                        false, A[m], false, B, (short)0, acc[m][j], false, false);
            }
        }
        __syncthreads();
    }

    /* C layout: VGPR r -> M = r + 8*hi ; N = ln */
    #pragma unroll
    for (int m = 0; m < 4; ++m)
        #pragma unroll
        for (int j = 0; j < 2; ++j) {
            int nt = wave + 8 * j;
            #pragma unroll
            for (int r = 0; r < 8; ++r) {
                int row = m * 16 + r + (hi ? 8 : 0);
                int col = nt * 16 + ln;
                atomicAdd(&accw[row * DD + col], acc[m][j][r]);
            }
        }
}

/* =================== S3: finalize stuff outputs =================== */
__global__ void __launch_bounds__(256)
stuff_final_kernel(const u32* __restrict__ scnt, const float* __restrict__ ssum,
                   const float* __restrict__ accw, float* __restrict__ out) {
    __shared__ float scale[64];
    int t = threadIdx.x;
    if (t < 64) {
        float scl = 0.0f;
        if (t < CS) {
            u32 cnt = scnt[t];
            float pix = fmaxf((float)cnt, 1.0f);
            float score = ssum[t] / pix;
            int keep = (score > SEM_THRES) && (cnt > 0u);
            out[OUT_SS + t]    = keep ? score : 0.0f;
            out[OUT_SKEEP + t] = keep ? 1.0f : 0.0f;
            scl = keep ? (1.0f / pix) : 0.0f;
        }
        scale[t] = scl;
    }
    __syncthreads();
    for (int i = t; i < CS * DD; i += 256) {
        int c = i >> 8;
        out[OUT_SK + i] = accw[i] * scale[c];
    }
}

/* =================== host-side launcher =================== */
extern "C" void kernel_launch(void* const* d_in, const int* in_sizes, int n_in,
                              void* d_out, int out_size, void* d_ws, size_t ws_size,
                              hipStream_t stream) {
    const float* thing = (const float*)d_in[0];   /* [1,80,256,512]  */
    const float* ksp   = (const float*)d_in[1];   /* [1,256,256,512] */
    const float* smap  = (const float*)d_in[2];   /* [1,54,256,512]  */
    float* out = (float*)d_out;

    char* ws = (char*)d_ws;
    u32*   w_hist  = (u32*)(ws + WS_HIST);
    u32*   w_cnt   = (u32*)(ws + WS_CNT);
    u32*   w_cut   = (u32*)(ws + WS_CUT);
    u32*   w_scnt  = (u32*)(ws + WS_SCNT);
    float* w_ssum  = (float*)(ws + WS_SSUM);
    int*   w_spat  = (int*)(ws + WS_SPAT);
    float* w_candv = (float*)(ws + WS_CANDV);
    int*   w_candi = (int*)(ws + WS_CANDI);
    float* w_acc   = (float*)(ws + WS_ACC);
    int*   w_cats  = (int*)(ws + WS_CATS);

    /* K0: zero hist/counters/stats/accumulators */
    zero_ws_kernel<<<(WS_ZERO_WORDS + 255) / 256, 256, 0, stream>>>(
        (u32*)ws, WS_ZERO_WORDS);

    /* things: histogram pass -> cutoff -> compaction pass */
    peaks_kernel<<<CT * 128, 256, 0, stream>>>(thing, w_hist, w_cut, w_cnt,
                                               w_candv, w_candi, 0);
    cutoff_kernel<<<1, 256, 0, stream>>>(w_hist, w_cut);
    peaks_kernel<<<CT * 128, 256, 0, stream>>>(thing, w_hist, w_cut, w_cnt,
                                               w_candv, w_candi, 1);
    select_kernel<<<1, 256, 0, stream>>>(w_cnt, w_candv, w_candi, out, w_spat);
    gather_kernel<<<KTOP, 256, 0, stream>>>(ksp, w_spat, out);

    /* stuff: argmax/stats -> TDM+WMMA GEMM -> finalize */
    stuff_argmax_kernel<<<HW / 256, 256, 0, stream>>>(smap, w_cats, w_scnt, w_ssum);
    stuff_gemm_kernel<<<S2_GRID, 256, 0, stream>>>(ksp, w_cats, w_acc);
    stuff_final_kernel<<<1, 256, 0, stream>>>(w_scnt, w_ssum, w_acc, out);
}
